// Q_NsNet2_npy_27565100106238
// MI455X (gfx1250) — matrix operand order repaired and checked by reading, not compiled, not run on hardware
//
#include <hip/hip_runtime.h>
#include <cmath>
#include <cstdint>
#include <cstddef>

typedef _Float16 v16h __attribute__((ext_vector_type(16)));
typedef _Float16 h8  __attribute__((ext_vector_type(8)));
typedef float    v8f __attribute__((ext_vector_type(8)));
typedef int      i4v __attribute__((ext_vector_type(4)));

// ---------------- problem dims ----------------
#define TSTEPS 200
#define BATCH  512
#define FDIM   257
#define HDIM   400
#define D2DIM  600

// ---------------- padded dims -----------------
#define FC1_KP 288    // 9*32
#define FC1_NP 448    // 7*64
#define GRU_KP 416    // 13*32
#define GRU_NP 1216   // 19*64
#define STST   448    // f16 stride for states / activations of width 400
#define FC2_NP 640    // 10*64
#define FC3_KP 608    // 19*32
#define FC4_NP 320    // 5*64
#define CHUNK_T 25
#define CHUNK_M (CHUNK_T*BATCH)   // 12800

#define LDA 40        // LDS stride (halfs) for 32-wide A tile rows
#define LDB 72        // LDS stride (halfs) for 64-wide B tile rows

// ---------------- async global->LDS (gfx1250) ---------------
#if defined(__AMDGCN__) && __has_builtin(__builtin_amdgcn_global_load_async_to_lds_b128)
#define HAVE_ASYNC_LDS 1
#else
#define HAVE_ASYNC_LDS 0
#endif

typedef __attribute__((address_space(1))) i4v* as1_i4p;
typedef __attribute__((address_space(3))) i4v* as3_i4p;

__device__ __forceinline__ void async_copy16(const _Float16* gsrc, _Float16* ldst)
{
#if HAVE_ASYNC_LDS
  __builtin_amdgcn_global_load_async_to_lds_b128((as1_i4p)gsrc, (as3_i4p)ldst, 0, 0);
#else
  *(uint4*)ldst = *(const uint4*)gsrc;
#endif
}

__device__ __forceinline__ void async_wait()
{
#if HAVE_ASYNC_LDS
#if __has_builtin(__builtin_amdgcn_s_wait_asynccnt)
  __builtin_amdgcn_s_wait_asynccnt(0);
#else
  asm volatile("s_wait_asynccnt 0x0" ::: "memory");
#endif
#endif
}

// ---------------- fake quantize (reciprocal multiply) ---------------
__device__ __forceinline__ float fqd(float v, float S, float iS, float Z) {
  float q = rintf(v * iS) + Z;
  q = fminf(fmaxf(q, 0.0f), 255.0f);
  return S * (q - Z);
}

// ======================================================================
// Generic 3-way WMMA GEMM:  C[M,N] = A[M,K](f16|fq(f32)) * B[K,N](f16)
// block tile 128x64, 8 waves (4x2), wave tile 32x32 (2x2 WMMA 16x16x32)
// ======================================================================
struct GemmDesc {
  const void*     A;
  const _Float16* B;
  void*           Out;
  const float*    bias;
  int   a_f32;            // 1: A is fp32, apply fq(aS,aZ) while staging
  float aS, aIS, aZ;
  int   Kact;             // real K (guard for fp32 path)
  int   Ksteps;           // padded K / 32
  int   strideA;          // elements
  int   strideB;          // elements (multiple of 64)
  int   Nact;
  int   strideO;
  int   epi;              // 0 f32 raw | 1 fc1 fq+bias+fq->f16 | 2 relu->f16 | 3 sigmoid->f32
  float e1S, e1IS, e1Z, e2S, e2IS, e2Z;
};

__global__ __launch_bounds__(256) void gemm3_kernel(GemmDesc ga, GemmDesc gb, GemmDesc gc)
{
  GemmDesc d = (blockIdx.z == 0) ? ga : ((blockIdx.z == 1) ? gb : gc);
  const int nTiles = (d.Nact + 63) >> 6;
  if ((int)blockIdx.x >= nTiles) return;

  __shared__ __align__(16) _Float16 sA[128 * LDA];
  __shared__ __align__(16) _Float16 sB[32 * LDB];

  const int tid  = threadIdx.x;
  const int lane = tid & 31;
  const int wave = tid >> 5;
  const int wm   = wave & 3;
  const int wn   = wave >> 2;
  const int m0   = blockIdx.y * 128;
  const int n0   = blockIdx.x * 64;

  v8f acc[2][2];
  #pragma unroll
  for (int i = 0; i < 2; ++i)
    #pragma unroll
    for (int j = 0; j < 2; ++j)
      #pragma unroll
      for (int k = 0; k < 8; ++k) acc[i][j][k] = 0.0f;

  for (int ks = 0; ks < d.Ksteps; ++ks) {
    const int kk = ks * 32;
    __syncthreads();
    { // stage A tile 128x32
      const int r = tid >> 1, hv = tid & 1;
      if (d.a_f32) {
        const float* Ap = (const float*)d.A + (size_t)(m0 + r) * d.strideA;
        #pragma unroll
        for (int j = 0; j < 16; ++j) {
          const int k = kk + hv * 16 + j;
          const float v = (k < d.Kact) ? Ap[k] : 0.0f;
          sA[r * LDA + hv * 16 + j] = (_Float16)fqd(v, d.aS, d.aIS, d.aZ);
        }
      } else {
        const _Float16* Ap = (const _Float16*)d.A + (size_t)(m0 + r) * d.strideA + kk + hv * 16;
        async_copy16(Ap,     &sA[r * LDA + hv * 16]);
        async_copy16(Ap + 8, &sA[r * LDA + hv * 16 + 8]);
      }
    }
    { // stage B tile 32x64
      const int r = tid >> 3, c = (tid & 7) * 8;
      const _Float16* Bp = d.B + (size_t)(kk + r) * d.strideB + n0 + c;
      async_copy16(Bp, &sB[r * LDB + c]);
    }
    async_wait();
    __syncthreads();

    v16h af[2];
    const int lg = lane >> 4;
    #pragma unroll
    for (int tm = 0; tm < 2; ++tm) {
      const int mrow = wm * 32 + tm * 16 + (lane & 15);
      h8 lo = *(h8*)&sA[mrow * LDA + lg * 8];
      h8 hi = *(h8*)&sA[mrow * LDA + 16 + lg * 8];
      #pragma unroll
      for (int i = 0; i < 8; ++i) { af[tm][i] = lo[i]; af[tm][i + 8] = hi[i]; }
    }
    #pragma unroll
    for (int tn = 0; tn < 2; ++tn) {
      const int nb = wn * 32 + tn * 16;
      v16h bf;
      #pragma unroll
      for (int i = 0; i < 8; ++i) { bf[i] = sB[(lg * 8 + i) * LDB + nb + (lane & 15)]; bf[i + 8] = sB[(16 + lg * 8 + i) * LDB + nb + (lane & 15)]; }
      #pragma unroll
      for (int tm = 0; tm < 2; ++tm)
        acc[tm][tn] = __builtin_amdgcn_wmma_f32_16x16x32_f16(
            false, af[tm], false, bf, (short)0, acc[tm][tn], false, false);
    }
  }

  const int lm = (lane >> 4) * 8;
  const int ln = lane & 15;
  #pragma unroll
  for (int tm = 0; tm < 2; ++tm)
    #pragma unroll
    for (int tn = 0; tn < 2; ++tn) {
      const int n = n0 + wn * 32 + tn * 16 + ln;
      if (n >= d.Nact) continue;
      const int mb = m0 + wm * 32 + tm * 16 + lm;
      #pragma unroll
      for (int v = 0; v < 8; ++v) {
        const int m = mb + v;
        const float x = acc[tm][tn][v];
        if (d.epi == 0) {
          ((float*)d.Out)[(size_t)m * d.strideO + n] = x;
        } else if (d.epi == 1) {
          float t = fqd(x, d.e1S, d.e1IS, d.e1Z) + d.bias[n];
          t = fqd(t, d.e2S, d.e2IS, d.e2Z);
          ((_Float16*)d.Out)[(size_t)m * d.strideO + n] = (_Float16)t;
        } else if (d.epi == 2) {
          const float t = x + d.bias[n];
          ((_Float16*)d.Out)[(size_t)m * d.strideO + n] = (_Float16)(t > 0.0f ? t : 0.0f);
        } else {
          const float t = x + d.bias[n];
          ((float*)d.Out)[(size_t)m * d.strideO + n] = 1.0f / (1.0f + __expf(-t));
        }
      }
    }
}

// ======================================================================
// Fused GRU: ix = xin@Wi^T (3 gate tiles per block, 12 wmma / K-step)
// then full gate math + fq in epilogue.  hh (h@Wh^T) precomputed (f32).
// ======================================================================
struct GruArgs {
  const _Float16* A;      // xin  [512 x STST]
  const _Float16* Bw;     // Wi^T [GRU_KP x GRU_NP]
  const float*    hh;     // [512 x GRU_NP]
  const _Float16* hcur;   // current (quantized) state [512 x STST]
  const float*    bi;     // [1200]
  const float*    bh;     // [1200]
  _Float16*       out1;   // raw new state (h1n / h2n)
  _Float16*       snext;  // fq('h*') new state (may alias hcur)
  int   use_fq;
  float Sa, ISa, Za, Sb, ISb, Zb, Scp, IScp, Zcp;
  float Sc, ISc, Zc, Sd, ISd, Zd, Sh, ISh, Zh;
};

__global__ __launch_bounds__(256) void gru_kernel(GruArgs d)
{
  __shared__ __align__(16) _Float16 sA[128 * LDA];
  __shared__ __align__(16) _Float16 sB[3][32 * LDB];

  const int tid  = threadIdx.x;
  const int lane = tid & 31;
  const int wave = tid >> 5;
  const int wm   = wave & 3;
  const int wn   = wave >> 2;
  const int m0   = blockIdx.y * 128;
  const int n0   = blockIdx.x * 64;

  v8f acc[3][2][2];
  #pragma unroll
  for (int g = 0; g < 3; ++g)
    #pragma unroll
    for (int i = 0; i < 2; ++i)
      #pragma unroll
      for (int j = 0; j < 2; ++j)
        #pragma unroll
        for (int k = 0; k < 8; ++k) acc[g][i][j][k] = 0.0f;

  for (int ks = 0; ks < 13; ++ks) {
    const int kk = ks * 32;
    __syncthreads();
    {
      const int r = tid >> 1, hv = tid & 1;
      const _Float16* Ap = d.A + (size_t)(m0 + r) * STST + kk + hv * 16;
      async_copy16(Ap,     &sA[r * LDA + hv * 16]);
      async_copy16(Ap + 8, &sA[r * LDA + hv * 16 + 8]);
    }
    {
      const int r = tid >> 3, c = (tid & 7) * 8;
      #pragma unroll
      for (int g = 0; g < 3; ++g) {
        const _Float16* Bp = d.Bw + (size_t)(kk + r) * GRU_NP + n0 + g * HDIM + c;
        async_copy16(Bp, &sB[g][r * LDB + c]);
      }
    }
    async_wait();
    __syncthreads();

    v16h af[2];
    const int lg = lane >> 4;
    #pragma unroll
    for (int tm = 0; tm < 2; ++tm) {
      const int mrow = wm * 32 + tm * 16 + (lane & 15);
      h8 lo = *(h8*)&sA[mrow * LDA + lg * 8];
      h8 hi = *(h8*)&sA[mrow * LDA + 16 + lg * 8];
      #pragma unroll
      for (int i = 0; i < 8; ++i) { af[tm][i] = lo[i]; af[tm][i + 8] = hi[i]; }
    }
    #pragma unroll
    for (int g = 0; g < 3; ++g) {
      #pragma unroll
      for (int tn = 0; tn < 2; ++tn) {
        const int nb = wn * 32 + tn * 16;
        v16h bf;
        #pragma unroll
        for (int i = 0; i < 8; ++i) { bf[i] = sB[g][(lg * 8 + i) * LDB + nb + (lane & 15)]; bf[i + 8] = sB[g][(16 + lg * 8 + i) * LDB + nb + (lane & 15)]; }
        #pragma unroll
        for (int tm = 0; tm < 2; ++tm)
          acc[g][tm][tn] = __builtin_amdgcn_wmma_f32_16x16x32_f16(
              false, af[tm], false, bf, (short)0, acc[g][tm][tn], false, false);
      }
    }
  }

  const int lm = (lane >> 4) * 8;
  const int ln = lane & 15;
  #pragma unroll
  for (int tm = 0; tm < 2; ++tm)
    #pragma unroll
    for (int tn = 0; tn < 2; ++tn) {
      const int n = n0 + wn * 32 + tn * 16 + ln;
      if (n >= HDIM) continue;
      const int mb = m0 + wm * 32 + tm * 16 + lm;
      #pragma unroll
      for (int v = 0; v < 8; ++v) {
        const int m = mb + v;
        const float iz  = acc[0][tm][tn][v] + d.bi[n];
        const float ir  = acc[1][tm][tn][v] + d.bi[n + HDIM];
        const float inn = acc[2][tm][tn][v] + d.bi[n + 2 * HDIM];
        const float* hr = d.hh + (size_t)m * GRU_NP;
        float zpre = iz + hr[n]        + d.bh[n];
        float rpre = ir + hr[n + HDIM] + d.bh[n + HDIM];
        float hnb  =      hr[n + 2*HDIM] + d.bh[n + 2*HDIM];
        if (d.use_fq) {
          zpre = fqd(zpre, d.Sa, d.ISa, d.Za);
          rpre = fqd(rpre, d.Sb, d.ISb, d.Zb);
          hnb  = fqd(hnb,  d.Scp, d.IScp, d.Zcp);
        }
        const float z = 1.0f / (1.0f + __expf(-zpre));
        const float r = 1.0f / (1.0f + __expf(-rpre));
        float npre = inn + r * hnb;
        if (d.use_fq) npre = fqd(npre, d.Sc, d.ISc, d.Zc);
        const float nn = tanhf(npre);
        const float hprev = (float)d.hcur[(size_t)m * STST + n];
        float hnew = (1.0f - z) * nn + z * hprev;
        if (d.use_fq) hnew = fqd(hnew, d.Sd, d.ISd, d.Zd);
        d.out1 [(size_t)m * STST + n] = (_Float16)hnew;
        d.snext[(size_t)m * STST + n] = (_Float16)fqd(hnew, d.Sh, d.ISh, d.Zh);
      }
    }
}

// ======================================================================
// Prep / utility kernels
// ======================================================================
__global__ void prep_weight(const float* src, _Float16* dst, int K, int N, int Kp, int Np,
                            int srcStride, int trans, int gateW,
                            float S0, float I0, float Z0,
                            float S1, float I1, float Z1,
                            float S2, float I2, float Z2)
{
  const int idx = blockIdx.x * blockDim.x + threadIdx.x;
  if (idx >= Kp * Np) return;
  const int k = idx / Np, n = idx % Np;
  float v = 0.0f;
  if (k < K && n < N) {
    const float w = trans ? src[(size_t)n * srcStride + k] : src[(size_t)k * srcStride + n];
    int g = n / gateW; if (g > 2) g = 2;
    const float S = (g == 0) ? S0 : ((g == 1) ? S1 : S2);
    const float I = (g == 0) ? I0 : ((g == 1) ? I1 : I2);
    const float Z = (g == 0) ? Z0 : ((g == 1) ? Z1 : Z2);
    v = fqd(w, S, I, Z);
  }
  dst[idx] = (_Float16)v;
}

__global__ void prep_bias(const float* src, float* dst, int n, int sliceW,
                          float S0, float I0, float Z0,
                          float S1, float I1, float Z1,
                          float S2, float I2, float Z2)
{
  const int i = blockIdx.x * blockDim.x + threadIdx.x;
  if (i >= n) return;
  int g = i / sliceW; if (g > 2) g = 2;
  const float S = (g == 0) ? S0 : ((g == 1) ? S1 : S2);
  const float I = (g == 0) ? I0 : ((g == 1) ? I1 : I2);
  const float Z = (g == 0) ? Z0 : ((g == 1) ? Z1 : Z2);
  dst[i] = fqd(src[i], S, I, Z);
}

__global__ void prep_state(const float* h, _Float16* s, float S, float I, float Z)
{
  const int i = blockIdx.x * blockDim.x + threadIdx.x;
  if (i >= BATCH * STST) return;
  const int b = i / STST, j = i % STST;
  const float v = (j < HDIM) ? fqd(h[(size_t)b * HDIM + j], S, I, Z) : 0.0f;
  s[i] = (_Float16)v;
}

__global__ void zero_kernel(uint32_t* p, long long n)
{
  long long i = (long long)blockIdx.x * blockDim.x + threadIdx.x;
  const long long stride = (long long)gridDim.x * blockDim.x;
  for (; i < n; i += stride) p[i] = 0u;
}

__global__ void finalize_kernel(const _Float16* h1last, const _Float16* h2last, float* out)
{
  const int i = blockIdx.x * blockDim.x + threadIdx.x;
  if (i >= BATCH * HDIM) return;
  const int b = i / HDIM, j = i % HDIM;
  out[i]                = (float)h1last[(size_t)b * STST + j];
  out[BATCH * HDIM + i] = (float)h2last[(size_t)b * STST + j];
}

// ======================================================================
// Host
// ======================================================================
struct SZ { float S, I, Z; };
static SZ mk(double mn, double mx) {
  const double S = (mx - mn) / 255.0;
  SZ r; r.S = (float)S; r.I = (float)(1.0 / S); r.Z = (float)nearbyint(-mn / S); return r;
}

extern "C" void kernel_launch(void* const* d_in, const int* in_sizes, int n_in,
                              void* d_out, int out_size, void* d_ws, size_t ws_size,
                              hipStream_t stream)
{
  (void)in_sizes; (void)n_in; (void)out_size; (void)ws_size;
  const float* x    = (const float*)d_in[0];
  const float* h1in = (const float*)d_in[1];
  const float* h2in = (const float*)d_in[2];
  const float* Wfc1 = (const float*)d_in[3];
  const float* bfc1 = (const float*)d_in[4];
  const float* Wi1  = (const float*)d_in[5];
  const float* Wh1  = (const float*)d_in[6];
  const float* b1   = (const float*)d_in[7];
  const float* Wi2  = (const float*)d_in[8];
  const float* Wh2  = (const float*)d_in[9];
  const float* b2   = (const float*)d_in[10];
  const float* Wfc2 = (const float*)d_in[11];
  const float* bfc2 = (const float*)d_in[12];
  const float* Wfc3 = (const float*)d_in[13];
  const float* bfc3 = (const float*)d_in[14];
  const float* Wfc4 = (const float*)d_in[15];
  const float* bfc4 = (const float*)d_in[16];
  float* out = (float*)d_out;

  // --- calibration ---
  const SZ cX   = mk(-0.0025095, 0.0022181);
  const SZ cH1  = mk(-0.002647488145157695, 0.0028339680284261703);
  const SZ cH2  = mk(-0.0031016061548143625, 0.0030250486452132463);
  const SZ cW166 = mk(-0.22075387835502625, 0.208940327167511);
  const SZ cF1M = mk(-0.00291599917, 0.0017367251);
  const SZ cF1B = mk(-0.48688140511512756, 0.5176185369491577);
  const SZ cF1A = mk(-0.48778465390205383, 0.5181604027748108);
  const SZ cWiz1 = mk(-0.43284985423088074, 0.46175122261047363);
  const SZ cWir1 = mk(-0.34401071071624756, 0.29191476106643677);
  const SZ cWin1 = mk(-0.3236880302429199, 0.39607325196266174);
  const SZ cWhz1 = mk(-1.8417714834213257, 1.7173254489898682);
  const SZ cWhr1 = mk(-1.1574513912200928, 1.0300449132919312);
  const SZ cWhn1 = mk(-0.7756922245025635, 0.9530389308929443);
  const SZ cBiz1 = mk(-0.5063393712043762, 0.36664387583732605);
  const SZ cBir1 = mk(-0.07920225709676743, 0.20611026883125305);
  const SZ cBin1 = mk(-0.5539973378181458, 0.17938342690467834);
  const SZ cBhz1 = mk(-0.5337516665458679, 0.4148772358894348);
  const SZ cBhr1 = mk(-0.07688436657190323, 0.14814253151416779);
  const SZ cBhn1 = mk(-0.7828555107116699, 0.9008108973503113);
  const SZ cG1a = mk(-0.6046768426895142, 0.8871182203292847);
  const SZ cG1b = mk(-0.07475128024816513, 0.14630259573459625);
  const SZ cG1cp = mk(-1.5660111904144287, 1.0454494953155518);
  const SZ cG1c = mk(-1.9779117107391357, 1.3700535297393799);
  const SZ cG1d = mk(-0.5529640316963196, 0.41507571935653687);
  const SZ cWiz2 = mk(-0.9102030992507935, 0.9408696889877319);
  const SZ cWir2 = mk(-0.9560997486114502, 0.6683358550071716);
  const SZ cWin2 = mk(-0.4721935987472534, 0.48561596870422363);
  const SZ cWhz2 = mk(-1.2992678880691528, 1.2991048097610474);
  const SZ cWhr2 = mk(-0.8318714499473572, 1.1085889339447021);
  const SZ cWhn2 = mk(-0.955470085144043, 1.046797513961792);
  const SZ cBiz2 = mk(-0.44805487990379333, 0.1560053527355194);
  const SZ cBir2 = mk(-0.08767592161893845, 0.11347303539514542);
  const SZ cBin2 = mk(-0.239909827709198, 0.12033259868621826);
  const SZ cBhz2 = mk(-0.43745461106300354, 0.12699371576309204);
  const SZ cBhr2 = mk(-0.09617264568805695, 0.07690174877643585);
  const SZ cBhn2 = mk(-0.17204178869724274, 0.19739042222499847);
  const SZ cW207 = mk(-1.3657219409942627, 1.158295750617981);
  const SZ cF2B = mk(-0.1750922054052353, 0.1385071724653244);
  const SZ cW208 = mk(-3.1666038036346436, 2.5026357173919678);
  const SZ cF3B = mk(-0.10188056528568268, 0.0899151861667633);
  const SZ cW209 = mk(-1.300571084022522, 1.928941249847412);
  const SZ cF4B = mk(-0.10699586570262909, 0.04597663879394531);
  const SZ zSZ = {1.0f, 1.0f, 0.0f};

  // --- workspace layout (256B aligned, +256B slack per buffer) ---
  char* base = (char*)d_ws;
  size_t off = 0;
  auto alloc = [&](size_t bytes) -> char* {
    char* p = base + off;
    off += ((bytes + 255) & ~(size_t)255) + 256;
    return p;
  };
  _Float16* Bfc1  = (_Float16*)alloc((size_t)FC1_KP * FC1_NP * 2);
  _Float16* BWi1  = (_Float16*)alloc((size_t)GRU_KP * GRU_NP * 2);
  _Float16* BWh1  = (_Float16*)alloc((size_t)GRU_KP * GRU_NP * 2);
  _Float16* BWi2  = (_Float16*)alloc((size_t)GRU_KP * GRU_NP * 2);
  _Float16* BWh2  = (_Float16*)alloc((size_t)GRU_KP * GRU_NP * 2);
  _Float16* Bfc2  = (_Float16*)alloc((size_t)GRU_KP * FC2_NP * 2);
  _Float16* Bfc3  = (_Float16*)alloc((size_t)FC3_KP * FC2_NP * 2);
  _Float16* Bfc4  = (_Float16*)alloc((size_t)FC3_KP * FC4_NP * 2);
  float* bfc1q = (float*)alloc(HDIM * 4);
  float* bi1q  = (float*)alloc(3 * HDIM * 4);
  float* bh1q  = (float*)alloc(3 * HDIM * 4);
  float* bi2q  = (float*)alloc(3 * HDIM * 4);
  float* bh2q  = (float*)alloc(3 * HDIM * 4);
  float* bfc2q = (float*)alloc(D2DIM * 4);
  float* bfc3q = (float*)alloc(D2DIM * 4);
  float* bfc4q = (float*)alloc(FDIM * 4);
  _Float16* aBuf  = (_Float16*)alloc((size_t)BATCH * STST * 2);
  _Float16* a1Buf = (_Float16*)alloc((size_t)BATCH * STST * 2);
  _Float16* s1    = (_Float16*)alloc((size_t)BATCH * STST * 2);
  _Float16* s2    = (_Float16*)alloc((size_t)BATCH * STST * 2);
  float* hh1 = (float*)alloc((size_t)BATCH * GRU_NP * 4);
  float* hh2 = (float*)alloc((size_t)BATCH * GRU_NP * 4);
  _Float16* y2 = (_Float16*)alloc((size_t)CHUNK_M * FC2_NP * 2);
  _Float16* y3 = (_Float16*)alloc((size_t)CHUNK_M * FC2_NP * 2);
  _Float16* h2all = (_Float16*)alloc((size_t)TSTEPS * BATCH * STST * 2);

  // --- zero pad-sensitive buffers ---
  zero_kernel<<<2048, 256, 0, stream>>>((uint32_t*)aBuf,  (long long)BATCH * STST / 2);
  zero_kernel<<<2048, 256, 0, stream>>>((uint32_t*)a1Buf, (long long)BATCH * STST / 2);
  zero_kernel<<<2048, 256, 0, stream>>>((uint32_t*)y2,    (long long)CHUNK_M * FC2_NP / 2);
  zero_kernel<<<2048, 256, 0, stream>>>((uint32_t*)y3,    (long long)CHUNK_M * FC2_NP / 2);
  zero_kernel<<<4096, 256, 0, stream>>>((uint32_t*)h2all, (long long)TSTEPS * BATCH * STST / 2);

  // --- quantize + pack weights (f16, padded) ---
  auto wgrid = [](size_t n) { return dim3((unsigned)((n + 255) / 256)); };
  prep_weight<<<wgrid((size_t)FC1_KP*FC1_NP),256,0,stream>>>(Wfc1, Bfc1, FDIM, HDIM, FC1_KP, FC1_NP, HDIM, 0, HDIM,
      cW166.S,cW166.I,cW166.Z, cW166.S,cW166.I,cW166.Z, cW166.S,cW166.I,cW166.Z);
  prep_weight<<<wgrid((size_t)GRU_KP*GRU_NP),256,0,stream>>>(Wi1, BWi1, HDIM, 3*HDIM, GRU_KP, GRU_NP, HDIM, 1, HDIM,
      cWiz1.S,cWiz1.I,cWiz1.Z, cWir1.S,cWir1.I,cWir1.Z, cWin1.S,cWin1.I,cWin1.Z);
  prep_weight<<<wgrid((size_t)GRU_KP*GRU_NP),256,0,stream>>>(Wh1, BWh1, HDIM, 3*HDIM, GRU_KP, GRU_NP, HDIM, 1, HDIM,
      cWhz1.S,cWhz1.I,cWhz1.Z, cWhr1.S,cWhr1.I,cWhr1.Z, cWhn1.S,cWhn1.I,cWhn1.Z);
  prep_weight<<<wgrid((size_t)GRU_KP*GRU_NP),256,0,stream>>>(Wi2, BWi2, HDIM, 3*HDIM, GRU_KP, GRU_NP, HDIM, 1, HDIM,
      cWiz2.S,cWiz2.I,cWiz2.Z, cWir2.S,cWir2.I,cWir2.Z, cWin2.S,cWin2.I,cWin2.Z);
  prep_weight<<<wgrid((size_t)GRU_KP*GRU_NP),256,0,stream>>>(Wh2, BWh2, HDIM, 3*HDIM, GRU_KP, GRU_NP, HDIM, 1, HDIM,
      cWhz2.S,cWhz2.I,cWhz2.Z, cWhr2.S,cWhr2.I,cWhr2.Z, cWhn2.S,cWhn2.I,cWhn2.Z);
  prep_weight<<<wgrid((size_t)GRU_KP*FC2_NP),256,0,stream>>>(Wfc2, Bfc2, HDIM, D2DIM, GRU_KP, FC2_NP, D2DIM, 0, D2DIM,
      cW207.S,cW207.I,cW207.Z, cW207.S,cW207.I,cW207.Z, cW207.S,cW207.I,cW207.Z);
  prep_weight<<<wgrid((size_t)FC3_KP*FC2_NP),256,0,stream>>>(Wfc3, Bfc3, D2DIM, D2DIM, FC3_KP, FC2_NP, D2DIM, 0, D2DIM,
      cW208.S,cW208.I,cW208.Z, cW208.S,cW208.I,cW208.Z, cW208.S,cW208.I,cW208.Z);
  prep_weight<<<wgrid((size_t)FC3_KP*FC4_NP),256,0,stream>>>(Wfc4, Bfc4, D2DIM, FDIM, FC3_KP, FC4_NP, FDIM, 0, FDIM,
      cW209.S,cW209.I,cW209.Z, cW209.S,cW209.I,cW209.Z, cW209.S,cW209.I,cW209.Z);

  // --- quantize biases (f32) ---
  prep_bias<<<wgrid(HDIM),256,0,stream>>>(bfc1, bfc1q, HDIM, HDIM,
      cF1B.S,cF1B.I,cF1B.Z, cF1B.S,cF1B.I,cF1B.Z, cF1B.S,cF1B.I,cF1B.Z);
  prep_bias<<<wgrid(3*HDIM),256,0,stream>>>(b1,          bi1q, 3*HDIM, HDIM,
      cBiz1.S,cBiz1.I,cBiz1.Z, cBir1.S,cBir1.I,cBir1.Z, cBin1.S,cBin1.I,cBin1.Z);
  prep_bias<<<wgrid(3*HDIM),256,0,stream>>>(b1 + 3*HDIM, bh1q, 3*HDIM, HDIM,
      cBhz1.S,cBhz1.I,cBhz1.Z, cBhr1.S,cBhr1.I,cBhr1.Z, cBhn1.S,cBhn1.I,cBhn1.Z);
  prep_bias<<<wgrid(3*HDIM),256,0,stream>>>(b2,          bi2q, 3*HDIM, HDIM,
      cBiz2.S,cBiz2.I,cBiz2.Z, cBir2.S,cBir2.I,cBir2.Z, cBin2.S,cBin2.I,cBin2.Z);
  prep_bias<<<wgrid(3*HDIM),256,0,stream>>>(b2 + 3*HDIM, bh2q, 3*HDIM, HDIM,
      cBhz2.S,cBhz2.I,cBhz2.Z, cBhr2.S,cBhr2.I,cBhr2.Z, cBhn2.S,cBhn2.I,cBhn2.Z);
  prep_bias<<<wgrid(D2DIM),256,0,stream>>>(bfc2, bfc2q, D2DIM, D2DIM,
      cF2B.S,cF2B.I,cF2B.Z, cF2B.S,cF2B.I,cF2B.Z, cF2B.S,cF2B.I,cF2B.Z);
  prep_bias<<<wgrid(D2DIM),256,0,stream>>>(bfc3, bfc3q, D2DIM, D2DIM,
      cF3B.S,cF3B.I,cF3B.Z, cF3B.S,cF3B.I,cF3B.Z, cF3B.S,cF3B.I,cF3B.Z);
  prep_bias<<<wgrid(FDIM),256,0,stream>>>(bfc4, bfc4q, FDIM, FDIM,
      cF4B.S,cF4B.I,cF4B.Z, cF4B.S,cF4B.I,cF4B.Z, cF4B.S,cF4B.I,cF4B.Z);

  // --- initial quantized states ---
  prep_state<<<wgrid((size_t)BATCH*STST),256,0,stream>>>(h1in, s1, cH1.S, cH1.I, cH1.Z);
  prep_state<<<wgrid((size_t)BATCH*STST),256,0,stream>>>(h2in, s2, cH2.S, cH2.I, cH2.Z);

  auto mkGemm = [](const void* A, int a_f32, SZ asz, int Kact, int Ksteps, int strideA,
                   const _Float16* B, int strideB, int Nact,
                   void* Out, int strideO, int epi, const float* bias, SZ e1, SZ e2) {
    GemmDesc d; d.A = A; d.B = B; d.Out = Out; d.bias = bias;
    d.a_f32 = a_f32; d.aS = asz.S; d.aIS = asz.I; d.aZ = asz.Z;
    d.Kact = Kact; d.Ksteps = Ksteps; d.strideA = strideA; d.strideB = strideB;
    d.Nact = Nact; d.strideO = strideO; d.epi = epi;
    d.e1S = e1.S; d.e1IS = e1.I; d.e1Z = e1.Z;
    d.e2S = e2.S; d.e2IS = e2.I; d.e2Z = e2.Z;
    return d;
  };

  // ================= recurrent phase =================
  for (int t = 0; t < TSTEPS; ++t) {
    GemmDesc g0 = mkGemm(x + (size_t)t * BATCH * FDIM, 1, cX, FDIM, 9, FDIM,
                         Bfc1, FC1_NP, HDIM, aBuf, STST, 1, bfc1q, cF1M, cF1A);
    GemmDesc g1 = mkGemm(s1, 0, zSZ, GRU_KP, 13, STST, BWh1, GRU_NP, 3 * HDIM,
                         hh1, GRU_NP, 0, nullptr, zSZ, zSZ);
    GemmDesc g2 = mkGemm(s2, 0, zSZ, GRU_KP, 13, STST, BWh2, GRU_NP, 3 * HDIM,
                         hh2, GRU_NP, 0, nullptr, zSZ, zSZ);
    gemm3_kernel<<<dim3(19, 4, 3), 256, 0, stream>>>(g0, g1, g2);

    GruArgs r1;
    r1.A = aBuf; r1.Bw = BWi1; r1.hh = hh1; r1.hcur = s1; r1.bi = bi1q; r1.bh = bh1q;
    r1.out1 = a1Buf; r1.snext = s1; r1.use_fq = 1;
    r1.Sa = cG1a.S; r1.ISa = cG1a.I; r1.Za = cG1a.Z;
    r1.Sb = cG1b.S; r1.ISb = cG1b.I; r1.Zb = cG1b.Z;
    r1.Scp = cG1cp.S; r1.IScp = cG1cp.I; r1.Zcp = cG1cp.Z;
    r1.Sc = cG1c.S; r1.ISc = cG1c.I; r1.Zc = cG1c.Z;
    r1.Sd = cG1d.S; r1.ISd = cG1d.I; r1.Zd = cG1d.Z;
    r1.Sh = cH1.S; r1.ISh = cH1.I; r1.Zh = cH1.Z;
    gru_kernel<<<dim3(7, 4, 1), 256, 0, stream>>>(r1);

    GruArgs r2;
    r2.A = a1Buf; r2.Bw = BWi2; r2.hh = hh2; r2.hcur = s2; r2.bi = bi2q; r2.bh = bh2q;
    r2.out1 = h2all + (size_t)t * BATCH * STST; r2.snext = s2; r2.use_fq = 0;
    r2.Sa = 1.f; r2.ISa = 1.f; r2.Za = 0.f;
    r2.Sb = 1.f; r2.ISb = 1.f; r2.Zb = 0.f;
    r2.Scp = 1.f; r2.IScp = 1.f; r2.Zcp = 0.f;
    r2.Sc = 1.f; r2.ISc = 1.f; r2.Zc = 0.f;
    r2.Sd = 1.f; r2.ISd = 1.f; r2.Zd = 0.f;
    r2.Sh = cH2.S; r2.ISh = cH2.I; r2.Zh = cH2.Z;
    gru_kernel<<<dim3(7, 4, 1), 256, 0, stream>>>(r2);
  }

  // ================= feed-forward phase (batched over time) =================
  for (int c = 0; c < TSTEPS / CHUNK_T; ++c) {
    const _Float16* Ah = h2all + (size_t)c * CHUNK_M * STST;
    GemmDesc f2 = mkGemm(Ah, 0, zSZ, GRU_KP, 13, STST, Bfc2, FC2_NP, D2DIM,
                         y2, FC2_NP, 2, bfc2q, zSZ, zSZ);
    gemm3_kernel<<<dim3(10, CHUNK_M / 128, 1), 256, 0, stream>>>(f2, f2, f2);
    GemmDesc f3 = mkGemm(y2, 0, zSZ, FC3_KP, 19, FC2_NP, Bfc3, FC2_NP, D2DIM,
                         y3, FC2_NP, 2, bfc3q, zSZ, zSZ);
    gemm3_kernel<<<dim3(10, CHUNK_M / 128, 1), 256, 0, stream>>>(f3, f3, f3);
    GemmDesc f4 = mkGemm(y3, 0, zSZ, FC3_KP, 19, FC2_NP, Bfc4, FC4_NP, FDIM,
                         out + (size_t)c * CHUNK_M * FDIM, FDIM, 3, bfc4q, zSZ, zSZ);
    gemm3_kernel<<<dim3(5, CHUNK_M / 128, 1), 256, 0, stream>>>(f4, f4, f4);
  }

  // final states h1f, h2f appended after masks
  finalize_kernel<<<(BATCH * HDIM + 255) / 256, 256, 0, stream>>>(
      a1Buf, h2all + (size_t)(TSTEPS - 1) * BATCH * STST,
      out + (size_t)TSTEPS * BATCH * FDIM);
}
